// Attention_76407468196387
// MI455X (gfx1250) — matrix operand order repaired and checked by reading, not compiled, not run on hardware
//
#include <hip/hip_runtime.h>
#include <hip/hip_bf16.h>
#include <stdint.h>

#define D_MODEL  1024
#define HEAD_DIM 128
#define SEQ      4096
#define NBATCH   4

typedef __attribute__((ext_vector_type(16))) __bf16 v16bf;
typedef __attribute__((ext_vector_type(8)))  float  v8f;

// ---- CDNA5 async global->LDS copy (ASYNCcnt) with portable fallback --------
#if __has_builtin(__builtin_amdgcn_global_load_async_to_lds_b128)
#define ASYNC_LDS 1
#else
#define ASYNC_LDS 0
#endif

#if ASYNC_LDS
#if __has_builtin(__builtin_amdgcn_s_wait_asynccnt)
#define WAIT_ASYNC() __builtin_amdgcn_s_wait_asynccnt(0)
#else
#define WAIT_ASYNC() asm volatile("s_wait_asynccnt 0" ::: "memory")
#endif
#else
#define WAIT_ASYNC() ((void)0)
#endif

// Builtin takes pointers to 16B int vectors in AS1 (global) and AS3 (LDS).
typedef int v4i_ __attribute__((vector_size(16)));
typedef __attribute__((address_space(1))) v4i_ gas_v4i;
typedef __attribute__((address_space(3))) v4i_ las_v4i;

static __device__ __forceinline__ void async_copy16(__bf16* lds, const __bf16* g) {
#if ASYNC_LDS
    // Flat LDS pointers carry the LDS byte offset in their low 32 bits.
    __builtin_amdgcn_global_load_async_to_lds_b128(
        (gas_v4i*)(uintptr_t)g,
        (las_v4i*)(uint32_t)(uintptr_t)lds, 0, 0);
#else
    *(uint4*)lds = *(const uint4*)g;
#endif
}

// Stage a (512/CHUNKS) x (CHUNKS*16B) tile, 16B per thread-slot, 256 threads.
template <int LDSTRIDE, int CHUNKS>
static __device__ __forceinline__ void stage_tile(__bf16* lds, const __bf16* g,
                                                  size_t gStrideHalves, int tid) {
#pragma unroll
    for (int it = 0; it < 2; ++it) {
        int c   = tid + it * 256;          // 0..511
        int row = c / CHUNKS;
        int cp  = c % CHUNKS;
        async_copy16(lds + row * LDSTRIDE + cp * 8,
                     g + (size_t)row * gStrideHalves + cp * 8);
    }
}

// 16 bf16 halves (8 VGPRs) viewed as two 16B chunks for b128 loads.
union FragB {
    v16bf  v;
    __bf16 h[16];
    uint4  q[2];
};

static __device__ __forceinline__ float red_max16(float x) {
    x = fmaxf(x, __shfl_xor(x, 1, 32));
    x = fmaxf(x, __shfl_xor(x, 2, 32));
    x = fmaxf(x, __shfl_xor(x, 4, 32));
    x = fmaxf(x, __shfl_xor(x, 8, 32));
    return x;
}
static __device__ __forceinline__ float red_sum16(float x) {
    x += __shfl_xor(x, 1, 32);
    x += __shfl_xor(x, 2, 32);
    x += __shfl_xor(x, 4, 32);
    x += __shfl_xor(x, 8, 32);
    return x;
}

// ---------------------------------------------------------------------------
// Kernel 0: Wq|Wk|Wv fp32 [1024,128] -> bf16 Wt[mat][col][k] (k contiguous).
// ---------------------------------------------------------------------------
__global__ void convert_weights(const float* __restrict__ Wq,
                                const float* __restrict__ Wk,
                                const float* __restrict__ Wv,
                                __bf16* __restrict__ Wt) {
    int idx = blockIdx.x * blockDim.x + threadIdx.x;
    if (idx >= 3 * HEAD_DIM * D_MODEL) return;
    int mat = idx / (HEAD_DIM * D_MODEL);
    int rem = idx % (HEAD_DIM * D_MODEL);
    int col = rem / D_MODEL;
    int k   = rem % D_MODEL;
    const float* W = (mat == 0) ? Wq : (mat == 1) ? Wk : Wv;
    Wt[idx] = (__bf16)W[k * HEAD_DIM + col];
}

// ---------------------------------------------------------------------------
// Kernel 1: QKV projection. Block = 128 rows (8 waves x 16 rows), W chunks
// double-buffered in LDS via async copies. Q pre-scaled; V stored transposed.
// ---------------------------------------------------------------------------
__global__ void __launch_bounds__(256)
qkv_project(const float* __restrict__ x, const __bf16* __restrict__ Wt,
            __bf16* __restrict__ Qb, __bf16* __restrict__ Kb,
            __bf16* __restrict__ Vt) {
    __shared__ __align__(16) __bf16 Ws[2][128 * 40];   // 128 cols x 32 k, rows padded to 80B

    const int tile  = blockIdx.x;
    const int b     = tile / (SEQ / 128);
    const int srow0 = (tile % (SEQ / 128)) * 128;
    const int tid   = threadIdx.x;
    const int w     = tid >> 5;
    const int lane  = tid & 31;
    const int lg    = lane >> 4, ln = lane & 15;
    const int mrow  = srow0 + w * 16 + ln;
    const float* xrow = x + ((size_t)b * SEQ + mrow) * D_MODEL;

    // Prime chunk 0 (mat 0, kb 0).
    stage_tile<40, 4>(&Ws[0][0], Wt, D_MODEL, tid);
    WAIT_ASYNC();
    __syncthreads();

    for (int mat = 0; mat < 3; ++mat) {
        v8f zero = {};
        v8f acc[8];
#pragma unroll
        for (int n = 0; n < 8; ++n) acc[n] = zero;

        for (int kb = 0; kb < D_MODEL; kb += 32) {
            const int ci  = (mat << 5) + (kb >> 5);
            const int cur = ci & 1;
            if (ci + 1 < 96) {  // stage next chunk (rolls over to next matrix)
                const int nci  = ci + 1;
                const int nmat = nci >> 5;
                const int nkb  = (nci & 31) << 5;
                stage_tile<40, 4>(&Ws[cur ^ 1][0],
                                  Wt + (size_t)nmat * HEAD_DIM * D_MODEL + nkb,
                                  D_MODEL, tid);
            }

            // A fragment: halves 0..15 <- x[kb+16lg..+15]
            FragB a;
            const float4* p0 = (const float4*)(xrow + kb + 16 * lg);
            const float4* p1 = (const float4*)(xrow + kb + 16 * lg + 8);
            float4 f0 = p0[0], f1 = p0[1], f2 = p1[0], f3 = p1[1];
            a.h[0]  = (__bf16)f0.x; a.h[1]  = (__bf16)f0.y;
            a.h[2]  = (__bf16)f0.z; a.h[3]  = (__bf16)f0.w;
            a.h[4]  = (__bf16)f1.x; a.h[5]  = (__bf16)f1.y;
            a.h[6]  = (__bf16)f1.z; a.h[7]  = (__bf16)f1.w;
            a.h[8]  = (__bf16)f2.x; a.h[9]  = (__bf16)f2.y;
            a.h[10] = (__bf16)f2.z; a.h[11] = (__bf16)f2.w;
            a.h[12] = (__bf16)f3.x; a.h[13] = (__bf16)f3.y;
            a.h[14] = (__bf16)f3.z; a.h[15] = (__bf16)f3.w;

            // Batch all 8 B-fragment loads (one clause, one wait), then 8 WMMAs
            // back-to-back so LDS latency overlaps matrix-op issue.
            FragB bfr[8];
#pragma unroll
            for (int n = 0; n < 8; ++n) {
                const uint4* wp =
                    (const uint4*)(&Ws[cur][0] + (n * 16 + ln) * 40 + 16 * lg);
                bfr[n].q[0] = wp[0]; bfr[n].q[1] = wp[1];
            }
#pragma unroll
            for (int n = 0; n < 8; ++n)
                acc[n] = __builtin_amdgcn_wmma_f32_16x16x32_bf16(
                    false, a.v, false, bfr[n].v, (short)0, acc[n], false, false);

            WAIT_ASYNC();       // our async stage for the next chunk is done
            __syncthreads();    // everyone done reading Ws[cur]
        }

        const float qscale = 0.08838834764831845f;   // 1/sqrt(128)
#pragma unroll
        for (int r = 0; r < 8; ++r) {
            int row = srow0 + w * 16 + r + 8 * lg;
#pragma unroll
            for (int n = 0; n < 8; ++n) {
                float vv  = acc[n][r];
                int   col = n * 16 + ln;
                if (mat == 0)
                    Qb[((size_t)b * SEQ + row) * HEAD_DIM + col] = (__bf16)(vv * qscale);
                else if (mat == 1)
                    Kb[((size_t)b * SEQ + row) * HEAD_DIM + col] = (__bf16)vv;
                else
                    Vt[((size_t)b * HEAD_DIM + col) * SEQ + row] = (__bf16)vv;
            }
        }
    }
}

// ---------------------------------------------------------------------------
// Kernel 2: causal flash attention. Block = 128 q-rows (8 waves x 16 rows).
// K/V chunks (32 keys) double-buffered in LDS via async copies; online
// softmax; P relayout via per-wave LDS bounce; O accumulated in fp32.
// ---------------------------------------------------------------------------
__global__ void __launch_bounds__(256)
flash_attn(const __bf16* __restrict__ Qb, const __bf16* __restrict__ Kb,
           const __bf16* __restrict__ Vt, float* __restrict__ out) {
    __shared__ __align__(16) __bf16 ldsP[8 * 16 * 40];   // per-wave 16x32 P (padded rows)
    __shared__ __align__(16) __bf16 Ks[2][32 * 136];     // 32 keys x 128 h (rows 272B)
    __shared__ __align__(16) __bf16 Vs[2][128 * 40];     // 128 h x 32 keys (rows 80B)

    const int tile = blockIdx.x;
    const int b    = tile / (SEQ / 128);
    const int q0   = (tile % (SEQ / 128)) * 128;
    const int tid  = threadIdx.x;
    const int w    = tid >> 5;
    const int lane = tid & 31;
    const int lg   = lane >> 4, ln = lane & 15;
    const int m0   = q0 + w * 16;

    const __bf16* Kbase = Kb + (size_t)b * SEQ * HEAD_DIM;
    const __bf16* Vbase = Vt + (size_t)b * HEAD_DIM * SEQ;

    // Q A-fragments for 4 chunks of 32 along head dim (Q already pre-scaled).
    FragB qf[4];
    {
        const __bf16* qrow = Qb + ((size_t)b * SEQ + m0 + ln) * HEAD_DIM;
#pragma unroll
        for (int c = 0; c < 4; ++c) {
            const uint4* p0 = (const uint4*)(qrow + c * 32 + 16 * lg);
            const uint4* p1 = (const uint4*)(qrow + c * 32 + 16 * lg + 8);
            qf[c].q[0] = p0[0];
            qf[c].q[1] = p1[0];
        }
    }

    float mrun[8], lrun[8];
    v8f zero = {};
    v8f o[8];
#pragma unroll
    for (int r = 0; r < 8; ++r) { mrun[r] = -3.0e38f; lrun[r] = 0.0f; }
#pragma unroll
    for (int ht = 0; ht < 8; ++ht) o[ht] = zero;

    __bf16* myP = ldsP + w * (16 * 40);
    const int qend = q0 + 128;

    // Prime key chunk 0.
    stage_tile<136, 16>(&Ks[0][0], Kbase, HEAD_DIM, tid);
    stage_tile<40, 4>(&Vs[0][0], Vbase, SEQ, tid);
    WAIT_ASYNC();
    __syncthreads();

    for (int kb = 0; kb < qend; kb += 32) {
        const int cur = (kb >> 5) & 1;
        if (kb + 32 < qend) {  // overlap next chunk's copy with this chunk's math
            stage_tile<136, 16>(&Ks[cur ^ 1][0], Kbase + (size_t)(kb + 32) * HEAD_DIM,
                                HEAD_DIM, tid);
            stage_tile<40, 4>(&Vs[cur ^ 1][0], Vbase + (kb + 32), SEQ, tid);
        }

        // S = Q K^T for 2 x 16 keys (batch the 4 K-fragment loads per tile).
        v8f st[2];
#pragma unroll
        for (int t = 0; t < 2; ++t) {
            const int n0 = kb + t * 16;
            const __bf16* krow = &Ks[cur][0] + (t * 16 + ln) * 136;
            FragB kf[4];
#pragma unroll
            for (int c = 0; c < 4; ++c) {
                const uint4* kp = (const uint4*)(krow + c * 32 + 16 * lg);
                kf[c].q[0] = kp[0]; kf[c].q[1] = kp[1];
            }
            v8f s = zero;
#pragma unroll
            for (int c = 0; c < 4; ++c)
                s = __builtin_amdgcn_wmma_f32_16x16x32_bf16(
                    false, qf[c].v, false, kf[c].v, (short)0, s, false, false);

            const int key = n0 + ln;
#pragma unroll
            for (int r = 0; r < 8; ++r) {
                int qrow = m0 + r + 8 * lg;
                if (key > qrow) s[r] = -3.0e38f;     // causal mask
            }
            st[t] = s;
        }

        // Online softmax update (each row lives in one 16-lane half).
        v8f sc;
#pragma unroll
        for (int r = 0; r < 8; ++r) {
            float rmax  = red_max16(fmaxf(st[0][r], st[1][r]));
            float mnew  = fmaxf(mrun[r], rmax);
            float alpha = __expf(mrun[r] - mnew);
            float p0    = __expf(st[0][r] - mnew);
            float p1    = __expf(st[1][r] - mnew);
            float rsum  = red_sum16(p0 + p1);
            lrun[r] = lrun[r] * alpha + rsum;
            mrun[r] = mnew;
            sc[r]   = alpha;
            int row = r + 8 * lg;
            myP[row * 40 + ln]      = (__bf16)p0;    // C layout -> LDS tile
            myP[row * 40 + 16 + ln] = (__bf16)p1;
        }
#pragma unroll
        for (int ht = 0; ht < 8; ++ht) o[ht] *= sc;

        asm volatile("s_wait_dscnt 0" ::: "memory");
        __builtin_amdgcn_wave_barrier();

        // Re-read P as a 16x32 A-fragment (2x ds_load_b128).
        FragB pf;
        {
            const uint4* pp0 = (const uint4*)(myP + ln * 40 + 16 * lg);
            const uint4* pp1 = (const uint4*)(myP + ln * 40 + 16 * lg + 8);
            pf.q[0] = pp0[0];
            pf.q[1] = pp1[0];
        }
        __builtin_amdgcn_wave_barrier();

        // O += P @ V; V-fragments loaded in batches of 4 to bound VGPR use.
#pragma unroll
        for (int hb = 0; hb < 2; ++hb) {
            FragB vf[4];
#pragma unroll
            for (int j = 0; j < 4; ++j) {
                const int ht = hb * 4 + j;
                const uint4* vp =
                    (const uint4*)(&Vs[cur][0] + (ht * 16 + ln) * 40 + 16 * lg);
                vf[j].q[0] = vp[0]; vf[j].q[1] = vp[1];
            }
#pragma unroll
            for (int j = 0; j < 4; ++j)
                o[hb * 4 + j] = __builtin_amdgcn_wmma_f32_16x16x32_bf16(
                    false, pf.v, false, vf[j].v, (short)0, o[hb * 4 + j],
                    false, false);
        }

        WAIT_ASYNC();       // next chunk's async copy has landed
        __syncthreads();    // all waves done with buffer `cur`
    }

    // Epilogue: divide by running softmax sum, store fp32.
#pragma unroll
    for (int r = 0; r < 8; ++r) {
        float inv = 1.0f / lrun[r];
        int   row = m0 + r + 8 * lg;
        float* orow = out + ((size_t)b * SEQ + row) * HEAD_DIM;
#pragma unroll
        for (int ht = 0; ht < 8; ++ht)
            orow[ht * 16 + ln] = o[ht][r] * inv;
    }
}

// ---------------------------------------------------------------------------
extern "C" void kernel_launch(void* const* d_in, const int* in_sizes, int n_in,
                              void* d_out, int out_size, void* d_ws, size_t ws_size,
                              hipStream_t stream) {
    (void)in_sizes; (void)n_in; (void)out_size; (void)ws_size;
    const float* x  = (const float*)d_in[0];
    const float* Wq = (const float*)d_in[1];
    const float* Wk = (const float*)d_in[2];
    const float* Wv = (const float*)d_in[3];
    float* out = (float*)d_out;

    char*  ws  = (char*)d_ws;
    __bf16* Wt = (__bf16*)ws;
    size_t off = (size_t)3 * HEAD_DIM * D_MODEL * sizeof(__bf16);     // 768 KB
    __bf16* Qb = (__bf16*)(ws + off); off += (size_t)NBATCH * SEQ * HEAD_DIM * 2;
    __bf16* Kb = (__bf16*)(ws + off); off += (size_t)NBATCH * SEQ * HEAD_DIM * 2;
    __bf16* Vt = (__bf16*)(ws + off);                                  // ~12.8 MB total

    convert_weights<<<(3 * HEAD_DIM * D_MODEL + 255) / 256, 256, 0, stream>>>(
        Wq, Wk, Wv, Wt);
    qkv_project<<<NBATCH * (SEQ / 128), 256, 0, stream>>>(x, Wt, Qb, Kb, Vt);
    flash_attn<<<NBATCH * (SEQ / 128), 256, 0, stream>>>(Qb, Kb, Vt, out);
}
